// ContrastiveLoss_66623532696170
// MI455X (gfx1250) — compile-verified, matrix operand order checked by reading, not run on hardware
//
#include <hip/hip_runtime.h>
#include <hip/hip_bf16.h>
#include <math.h>

typedef _Float16 v16h __attribute__((ext_vector_type(16)));
typedef _Float16 v8h  __attribute__((ext_vector_type(8)));
typedef float    v8f  __attribute__((ext_vector_type(8)));
typedef float    f4   __attribute__((ext_vector_type(4)));

#define DIM   256
#define TILE  16
#define KSTEP 32
#define EPS   1e-8f
// 0.5 * log2(e): folded into the stored A operand so the WMMA accumulator
// holds sim * log2(e) / 2 and the epilogue is a bare v_exp_f32 (exp2).
#define HALF_LOG2E 0.72134752044448170f

// ---------------------------------------------------------------- init ------
__global__ void cl_init_kernel(float* posAccum) {
    if (threadIdx.x == 0) posAccum[0] = 0.0f;
}

// ----------------------------------------------- normalize pos (+pos_sim) ---
// one wave (32 lanes) per row; 8 f32 per lane (DIM=256)
__global__ __launch_bounds__(256) void cl_norm_pos_kernel(
    const float* __restrict__ pk, const float* __restrict__ pv,
    _Float16* __restrict__ pkh, float* __restrict__ posAccum, int n)
{
    int wave = threadIdx.x >> 5;
    int lane = threadIdx.x & 31;
    int row  = blockIdx.x * 8 + wave;
    if (row >= n) return;

    const float* kp = pk + (size_t)row * DIM + lane * 8;
    const float* vp = pv + (size_t)row * DIM + lane * 8;

    float xk[8], xv[8];
    f4 k0 = *(const f4*)(kp);     f4 k1 = *(const f4*)(kp + 4);
    f4 v0 = *(const f4*)(vp);     f4 v1 = *(const f4*)(vp + 4);
    #pragma unroll
    for (int i = 0; i < 4; ++i) { xk[i] = k0[i]; xk[4+i] = k1[i];
                                  xv[i] = v0[i]; xv[4+i] = v1[i]; }

    float ssk = 0.f, ssv = 0.f, dot = 0.f;
    #pragma unroll
    for (int i = 0; i < 8; ++i) {
        ssk += xk[i] * xk[i];
        ssv += xv[i] * xv[i];
        dot += xk[i] * xv[i];
    }
    #pragma unroll
    for (int msk = 1; msk < 32; msk <<= 1) {
        ssk += __shfl_xor(ssk, msk, 32);
        ssv += __shfl_xor(ssv, msk, 32);
        dot += __shfl_xor(dot, msk, 32);
    }
    float invk = 1.0f / fmaxf(sqrtf(ssk), EPS);
    float invv = 1.0f / fmaxf(sqrtf(ssv), EPS);

    // store normalized pos_key * (log2e/2) as f16 (16B vector store)
    float s = invk * HALF_LOG2E;
    v8h h;
    #pragma unroll
    for (int i = 0; i < 8; ++i) h[i] = (_Float16)(xk[i] * s);
    *(v8h*)(pkh + (size_t)row * DIM + lane * 8) = h;

    if (lane == 0) atomicAdd(posAccum, dot * invk * invv);
}

// ------------------------------------------------------- normalize neg ------
__global__ __launch_bounds__(256) void cl_norm_neg_kernel(
    const float* __restrict__ nv, _Float16* __restrict__ nvh, int m)
{
    int wave = threadIdx.x >> 5;
    int lane = threadIdx.x & 31;
    int row  = blockIdx.x * 8 + wave;
    if (row >= m) return;

    const float* p = nv + (size_t)row * DIM + lane * 8;
    float x[8];
    f4 a0 = *(const f4*)(p); f4 a1 = *(const f4*)(p + 4);
    #pragma unroll
    for (int i = 0; i < 4; ++i) { x[i] = a0[i]; x[4+i] = a1[i]; }

    float ss = 0.f;
    #pragma unroll
    for (int i = 0; i < 8; ++i) ss += x[i] * x[i];
    #pragma unroll
    for (int msk = 1; msk < 32; msk <<= 1) ss += __shfl_xor(ss, msk, 32);
    float inv = 1.0f / fmaxf(sqrtf(ss), EPS);

    v8h h;
    #pragma unroll
    for (int i = 0; i < 8; ++i) h[i] = (_Float16)(x[i] * inv);
    *(v8h*)(nvh + (size_t)row * DIM + lane * 8) = h;
}

// ----------------------------------------- WMMA GEMM + fused logsumexp ------
// Fragment layout per CDNA5 ISA 7.12.2 (16-bit operand, 16x32 K-slice):
//   lanes 0-15 : K = k0+0..7  and  k0+16..23
//   lanes 16-31: K = k0+8..15 and  k0+24..31
__device__ __forceinline__ void cl_load_bfrag(
    const _Float16* __restrict__ brow, int khalf, v16h* b)
{
    #pragma unroll
    for (int kt = 0; kt < 8; ++kt) {
        v8h lo = *(const v8h*)(brow + kt * KSTEP + khalf);
        v8h hi = *(const v8h*)(brow + kt * KSTEP + khalf + 16);
        #pragma unroll
        for (int i = 0; i < 8; ++i) { b[kt][i] = lo[i]; b[kt][8 + i] = hi[i]; }
    }
}

__device__ __forceinline__ void cl_tile_mma_exp(
    const v16h* __restrict__ afr, const v16h* __restrict__ b, float* acc)
{
    v8f c = {0.f, 0.f, 0.f, 0.f, 0.f, 0.f, 0.f, 0.f};
    #pragma unroll
    for (int kt = 0; kt < 8; ++kt)
        c = __builtin_amdgcn_wmma_f32_16x16x32_f16(
                false, afr[kt], false, b[kt], (short)0, c, false, false);
    // accumulator already holds sim * log2e/2 -> bare exp2 epilogue
    #pragma unroll
    for (int r = 0; r < 8; ++r) acc[r] += __builtin_amdgcn_exp2f(c[r]);
}

// workgroup = 8 waves, owns 16 rows; waves split the m/16 column tiles.
// B fragments are double-buffered in registers so the 16 b128 loads for the
// next tile are in flight during the current tile's 8 WMMAs.
__global__ __launch_bounds__(256) void cl_gemm_lse_kernel(
    const _Float16* __restrict__ A, const _Float16* __restrict__ B,
    float* __restrict__ rowLse, int n, int m)
{
    __shared__ float srow[TILE];
    int tid  = threadIdx.x;
    int wave = tid >> 5;
    int lane = tid & 31;
    int rbase = blockIdx.x * TILE;

    if (tid < TILE) srow[tid] = 0.0f;
    __syncthreads();

    int lr    = lane & 15;          // row/col within tile
    int khalf = (lane >> 4) * 8;    // lane-group K offset

    // preload the wave's A fragments for all 8 K-slices (reused per tile)
    v16h afr[8];
    {
        const _Float16* arow = A + (size_t)(rbase + lr) * DIM;
        #pragma unroll
        for (int kt = 0; kt < 8; ++kt) {
            v8h lo = *(const v8h*)(arow + kt * KSTEP + khalf);
            v8h hi = *(const v8h*)(arow + kt * KSTEP + khalf + 16);
            #pragma unroll
            for (int i = 0; i < 8; ++i) { afr[kt][i] = lo[i]; afr[kt][8+i] = hi[i]; }
        }
    }

    float acc[8];
    #pragma unroll
    for (int r = 0; r < 8; ++r) acc[r] = 0.0f;

    int ntiles = m / TILE;
    v16h b0[8], b1[8];

    int j = wave;
    if (j < ntiles)
        cl_load_bfrag(B + (size_t)(j * TILE + lr) * DIM, khalf, b0);

    for (; j < ntiles; j += 16) {
        int j1 = j + 8;
        if (j1 < ntiles)
            cl_load_bfrag(B + (size_t)(j1 * TILE + lr) * DIM, khalf, b1);
        cl_tile_mma_exp(afr, b0, acc);

        if (j1 < ntiles) {
            int j2 = j + 16;
            if (j2 < ntiles)
                cl_load_bfrag(B + (size_t)(j2 * TILE + lr) * DIM, khalf, b0);
            cl_tile_mma_exp(afr, b1, acc);
        }
    }

    // reduce over the 16 lanes of each half-group (columns of the tile)
    #pragma unroll
    for (int r = 0; r < 8; ++r) {
        float v = acc[r];
        #pragma unroll
        for (int msk = 1; msk < 16; msk <<= 1) v += __shfl_xor(v, msk, 32);
        acc[r] = v;
    }
    if (lr == 0) {
        int rowoff = (lane >> 4) * 8;   // lanes 0-15 -> rows 0-7, 16-31 -> 8-15
        #pragma unroll
        for (int r = 0; r < 8; ++r) atomicAdd(&srow[rowoff + r], acc[r]);
    }
    __syncthreads();
    // logsumexp(sim/2): no shift needed, |sim| <= 1 so the sum is in range
    if (tid < TILE) rowLse[rbase + tid] = __logf(srow[tid]);
}

// ---------------------------------------------------------------- final -----
__global__ __launch_bounds__(256) void cl_combine_kernel(
    const float* __restrict__ rowLse, const float* __restrict__ posAccum,
    float* __restrict__ out, int n)
{
    __shared__ float s[256];
    float v = 0.0f;
    for (int i = threadIdx.x; i < n; i += 256) v += rowLse[i];
    s[threadIdx.x] = v;
    __syncthreads();
    for (int w = 128; w > 0; w >>= 1) {
        if (threadIdx.x < w) s[threadIdx.x] += s[threadIdx.x + w];
        __syncthreads();
    }
    if (threadIdx.x == 0) {
        float negMean = s[0] / (float)n;                    // mean logsumexp
        float posLoss = -posAccum[0] / (2.0f * (float)n);   // -mean(pos_sim)/TEMP
        out[0] = posLoss + negMean;
    }
}

// ---------------------------------------------------------------------------
extern "C" void kernel_launch(void* const* d_in, const int* in_sizes, int n_in,
                              void* d_out, int out_size, void* d_ws, size_t ws_size,
                              hipStream_t stream) {
    const float* pos_key   = (const float*)d_in[0];
    const float* pos_value = (const float*)d_in[1];
    const float* neg_value = (const float*)d_in[2];

    int n = in_sizes[0] / DIM;   // 8192
    int m = in_sizes[2] / DIM;   // 8192

    char* ws = (char*)d_ws;
    float* posAccum = (float*)ws;                     // 1 float
    float* rowLse   = (float*)(ws + 256);             // n floats
    size_t off = 256 + (size_t)n * sizeof(float);
    off = (off + 255) & ~(size_t)255;
    _Float16* PK = (_Float16*)(ws + off);             // n*DIM f16 (pre-scaled)
    off += (size_t)n * DIM * sizeof(_Float16);
    _Float16* NV = (_Float16*)(ws + off);             // m*DIM f16

    cl_init_kernel<<<1, 32, 0, stream>>>(posAccum);
    cl_norm_pos_kernel<<<(n + 7) / 8, 256, 0, stream>>>(pos_key, pos_value,
                                                        PK, posAccum, n);
    cl_norm_neg_kernel<<<(m + 7) / 8, 256, 0, stream>>>(neg_value, NV, m);
    cl_gemm_lse_kernel<<<n / TILE, 256, 0, stream>>>(PK, NV, rowLse, n, m);
    cl_combine_kernel<<<1, 256, 0, stream>>>(rowLse, posAccum, (float*)d_out, n);
}
